// DoubleConv_24111946400090
// MI455X (gfx1250) — 8-bit matrix operand order repaired and checked by reading, compiled, not run on hardware
//
#include <hip/hip_runtime.h>
#include <hip/hip_bf16.h>

typedef __attribute__((ext_vector_type(8))) int v8i;
typedef __attribute__((ext_vector_type(4))) int v4i;
typedef __attribute__((ext_vector_type(4))) unsigned v4u;

#define EPSQ 1e-5f
#define EPSBN 1e-5f

#if defined(__has_builtin)
#if __has_builtin(__builtin_amdgcn_tensor_load_to_lds)
#define HAVE_TDM 1
#endif
#endif

__device__ __forceinline__ void tdm_load_to_lds(const void* gaddr, unsigned ldsOff,
                                                unsigned bytes) {
#ifdef HAVE_TDM
  const unsigned rows = bytes >> 13;
  const unsigned td0 = 2048;
  const unsigned long long ga = (unsigned long long)(size_t)gaddr;
  v4u g0;
  g0.x = 1u;
  g0.y = ldsOff;
  g0.z = (unsigned)ga;
  g0.w = (unsigned)(ga >> 32) | 0x80000000u;
  v8i g1;
  g1[0] = (int)(2u << 16);
  g1[1] = (int)((td0 & 0xFFFFu) << 16);
  g1[2] = (int)((td0 >> 16) | ((rows & 0xFFFFu) << 16));
  g1[3] = (int)((rows >> 16) | ((td0 & 0xFFFFu) << 16));
  g1[4] = (int)(rows & 0xFFFFu);
  g1[5] = (int)td0;
  g1[6] = 0;
  g1[7] = 0;
  v4i z4 = {};
#if __clang_major__ >= 23
  v8i z8 = {};
  __builtin_amdgcn_tensor_load_to_lds(g0, g1, z4, z4, z8, 0);
#else
  __builtin_amdgcn_tensor_load_to_lds(g0, g1, z4, z4, 0);
#endif
  __builtin_amdgcn_s_wait_tensorcnt(0);
#else
  (void)gaddr; (void)ldsOff; (void)bytes;
#endif
}

__global__ __launch_bounds__(256) void weight_beta(const float* __restrict__ w1, int n1,
                                                   const float* __restrict__ w2, int n2,
                                                   float* __restrict__ betaB) {
  const float* w = blockIdx.x ? w2 : w1;
  const int n = blockIdx.x ? n2 : n1;
  float s = 0.f;
  for (int i = threadIdx.x; i < n; i += 256) s += fabsf(w[i]);
  __shared__ float red[256];
  red[threadIdx.x] = s;
  __syncthreads();
  for (int off = 128; off > 0; off >>= 1) {
    if (threadIdx.x < off) red[threadIdx.x] += red[threadIdx.x + off];
    __syncthreads();
  }
  if (threadIdx.x == 0) betaB[blockIdx.x] = fmaxf(red[0] / (float)n, EPSQ);
}

__global__ __launch_bounds__(256) void weight_pack(const float* __restrict__ w,
                                                   const float* __restrict__ betaP,
                                                   signed char* __restrict__ bpk,
                                                   int CIN, int KB, int total) {
  int idx = blockIdx.x * 256 + threadIdx.x;
  if (idx >= total) return;
  const float invb = 1.0f / betaP[0];
  const int j = idx & 31;
  const int lane = (idx >> 5) & 31;
  const int t = (idx >> 10) & 7;
  const int tapkb = idx >> 13;
  const int tap = tapkb / KB, kb = tapkb % KB;
  const int half = lane >> 4, ncol = lane & 15;
  const int v = j >> 2, b = j & 3;
  const int k64 = half * 8 + (j >> 3) * 16 + (j & 7);
  const int cin = kb * 64 + k64;
  const int cout = t * 16 + ncol;
  const int kh = tap / 3, kw = tap % 3;
  float wv = w[(((size_t)cout * CIN + cin) * 3 + kh) * 3 + kw];
  float q = rintf(wv * invb);
  q = fminf(fmaxf(q, -1.f), 1.f);
  bpk[idx] = (signed char)q;
}

__global__ __launch_bounds__(256) void absmax_x(const float* __restrict__ x,
                                                unsigned* __restrict__ gammaBits) {
  const int n = blockIdx.x >> 12;
  const int bi = blockIdx.x & 4095;
  const size_t base = (size_t)n * 4194304 + (size_t)bi * 1024 + threadIdx.x * 4;
  const float4 v = *(const float4*)(x + base);
  float m = fmaxf(fmaxf(fabsf(v.x), fabsf(v.y)), fmaxf(fabsf(v.z), fabsf(v.w)));
  __shared__ float red[256];
  red[threadIdx.x] = m;
  __syncthreads();
  for (int off = 128; off > 0; off >>= 1) {
    if (threadIdx.x < off) red[threadIdx.x] = fmaxf(red[threadIdx.x], red[threadIdx.x + off]);
    __syncthreads();
  }
  if (threadIdx.x == 0) atomicMax(&gammaBits[n], __float_as_uint(red[0]));
}

__global__ __launch_bounds__(256) void quant_x(const float* __restrict__ x,
                                               const unsigned* __restrict__ gammaBits,
                                               signed char* __restrict__ xi) {
  const size_t idx = (size_t)blockIdx.x * 256 + threadIdx.x;
  const int n = (int)(idx >> 22);
  const int c = (int)(idx >> 16) & 63;
  const int yx = (int)idx & 65535;
  const float g = fmaxf(__uint_as_float(gammaBits[n]), EPSQ);
  const float sc = 127.0f / g;
  float v = rintf(x[idx] * sc);
  v = fminf(fmaxf(v, -128.0f), 127.0f);
  xi[((size_t)n << 22) + (size_t)yx * 64 + c] = (signed char)v;
}

template <int CIN>
__global__ __launch_bounds__(256) void conv_wmma(const signed char* __restrict__ xi,
                                                 const signed char* __restrict__ bpk,
                                                 const float* __restrict__ betaP,
                                                 const unsigned* __restrict__ gammaBits,
                                                 float* __restrict__ yout,
                                                 float* __restrict__ psum,
                                                 float* __restrict__ psq) {
  constexpr int KB = CIN / 64;
  constexpr unsigned BPKB = 9u * KB * 8192u;
  extern __shared__ char cv_smem[];
  signed char* sB = (signed char*)cv_smem;
  float* lsum = (float*)(cv_smem + BPKB);
  float* lsq = lsum + 8 * 128;

  const int tid = threadIdx.x;
  const int lane = tid & 31;
  const int wave = tid >> 5;
  const int half = lane >> 4;
  const int nl = lane & 15;

#ifdef HAVE_TDM
  if (tid < 32) {
    tdm_load_to_lds(bpk, 0u, BPKB);
  }
#else
  for (unsigned o = (unsigned)tid * 16u; o < BPKB; o += 256u * 16u)
    *(v4i*)(sB + o) = *(const v4i*)(bpk + o);
#endif
  __syncthreads();

  const int tile = blockIdx.x * 8 + wave;
  const int pixbase = tile << 4;
  const int n = pixbase >> 16;
  const int yy = (pixbase >> 8) & 255;
  const int xx0 = pixbase & 255;

  v8i zero = {};
  v8i acc[8];
#pragma unroll
  for (int t = 0; t < 8; ++t) acc[t] = zero;

  for (int tap = 0; tap < 9; ++tap) {
    const int dy = tap / 3 - 1, dx = tap % 3 - 1;
    const int yq = yy + dy;
    const int xq = xx0 + nl + dx;
    const bool inb = ((unsigned)yq < 256u) && ((unsigned)xq < 256u);
#pragma unroll
    for (int kb = 0; kb < KB; ++kb) {
      union { v8i v; unsigned long long u[4]; } A;
      if (inb) {
        const unsigned long long* p = (const unsigned long long*)(
            xi + ((size_t)((n * 256 + yq) * 256 + xq)) * CIN + kb * 64 + half * 8);
        A.u[0] = p[0];
        A.u[1] = p[2];
        A.u[2] = p[4];
        A.u[3] = p[6];
      } else {
        A.u[0] = A.u[1] = A.u[2] = A.u[3] = 0ull;
      }
      const signed char* bb = sB + (unsigned)(tap * KB + kb) * 8192u + lane * 32;
#pragma unroll
      for (int t = 0; t < 8; ++t) {
        union { v8i v; v4i h[2]; } B;
        const v4i* bp = (const v4i*)(bb + t * 1024);
        B.h[0] = bp[0];
        B.h[1] = bp[1];
        acc[t] = __builtin_amdgcn_wmma_i32_16x16x64_iu8(true, A.v, true, B.v, acc[t],
                                                        false, false);
      }
    }
  }

  const float gm = fmaxf(__uint_as_float(gammaBits[n]), EPSQ);
  const float s = betaP[0] * gm * (1.0f / 127.0f);

#pragma unroll
  for (int t = 0; t < 8; ++t) {
    float ps = 0.f, pq = 0.f;
    const int c = t * 16 + nl;
#pragma unroll
    for (int i = 0; i < 8; ++i) {
      const int m = i + half * 8;
      const float y = (float)acc[t][i] * s;
      yout[((size_t)(pixbase + m)) * 128 + c] = y;
      ps += y;
      pq += y * y;
    }
    ps += __shfl_xor(ps, 16, 32);
    pq += __shfl_xor(pq, 16, 32);
    if (half == 0) { lsum[wave * 128 + c] = ps; lsq[wave * 128 + c] = pq; }
  }
  __syncthreads();
  if (tid < 128) {
    float s1 = 0.f, s2 = 0.f;
#pragma unroll
    for (int w = 0; w < 8; ++w) { s1 += lsum[w * 128 + tid]; s2 += lsq[w * 128 + tid]; }
    psum[(size_t)blockIdx.x * 128 + tid] = s1;
    psq[(size_t)blockIdx.x * 128 + tid] = s2;
  }
}

__global__ __launch_bounds__(256) void bn_finalize(const float* __restrict__ psum,
                                                   const float* __restrict__ psq,
                                                   const float* __restrict__ gamma,
                                                   const float* __restrict__ beta,
                                                   float* __restrict__ scl,
                                                   float* __restrict__ bias,
                                                   int nblocks, float invCount) {
  __shared__ float ss[256], sq[256];
  const int c = blockIdx.x;
  const int tid = threadIdx.x;
  float a = 0.f, b = 0.f;
  for (int i = tid; i < nblocks; i += 256) {
    a += psum[(size_t)i * 128 + c];
    b += psq[(size_t)i * 128 + c];
  }
  ss[tid] = a; sq[tid] = b;
  __syncthreads();
  for (int off = 128; off > 0; off >>= 1) {
    if (tid < off) { ss[tid] += ss[tid + off]; sq[tid] += sq[tid + off]; }
    __syncthreads();
  }
  if (tid == 0) {
    const float mean = ss[0] * invCount;
    const float var = sq[0] * invCount - mean * mean;
    const float sc = gamma[c] * rsqrtf(var + EPSBN);
    scl[c] = sc;
    bias[c] = beta[c] - mean * sc;
  }
}

__global__ __launch_bounds__(256) void bnrelu_absmax(float* __restrict__ y,
                                                     const float* __restrict__ scl,
                                                     const float* __restrict__ bias,
                                                     unsigned* __restrict__ gammaBits) {
  const int n = blockIdx.x >> 13;
  const int bi = blockIdx.x & 8191;
  const size_t base = (size_t)n * 8388608 + (size_t)bi * 1024 + threadIdx.x * 4;
  const int c0 = (threadIdx.x * 4) & 127;
  float4 v = *(float4*)(y + base);
  v.x = fmaxf(v.x * scl[c0 + 0] + bias[c0 + 0], 0.f);
  v.y = fmaxf(v.y * scl[c0 + 1] + bias[c0 + 1], 0.f);
  v.z = fmaxf(v.z * scl[c0 + 2] + bias[c0 + 2], 0.f);
  v.w = fmaxf(v.w * scl[c0 + 3] + bias[c0 + 3], 0.f);
  *(float4*)(y + base) = v;
  float m = fmaxf(fmaxf(v.x, v.y), fmaxf(v.z, v.w));
  __shared__ float red[256];
  red[threadIdx.x] = m;
  __syncthreads();
  for (int off = 128; off > 0; off >>= 1) {
    if (threadIdx.x < off) red[threadIdx.x] = fmaxf(red[threadIdx.x], red[threadIdx.x + off]);
    __syncthreads();
  }
  if (threadIdx.x == 0) atomicMax(&gammaBits[n], __float_as_uint(red[0]));
}

__global__ __launch_bounds__(256) void quant_h(const float* __restrict__ h,
                                               const unsigned* __restrict__ gammaBits,
                                               signed char* __restrict__ xi2) {
  const size_t idx4 = ((size_t)blockIdx.x * 256 + threadIdx.x) * 4;
  const int n = (int)(idx4 >> 23);
  const float g = fmaxf(__uint_as_float(gammaBits[n]), EPSQ);
  const float sc = 127.0f / g;
  const float4 v = *(const float4*)(h + idx4);
  int b0 = (int)fminf(fmaxf(rintf(v.x * sc), -128.f), 127.f);
  int b1 = (int)fminf(fmaxf(rintf(v.y * sc), -128.f), 127.f);
  int b2 = (int)fminf(fmaxf(rintf(v.z * sc), -128.f), 127.f);
  int b3 = (int)fminf(fmaxf(rintf(v.w * sc), -128.f), 127.f);
  const int packed = (b0 & 255) | ((b1 & 255) << 8) | ((b2 & 255) << 16) | ((b3 & 255) << 24);
  *(int*)(xi2 + idx4) = packed;
}

__global__ __launch_bounds__(256) void bnrelu_out(const float* __restrict__ y,
                                                  const float* __restrict__ scl,
                                                  const float* __restrict__ bias,
                                                  float* __restrict__ out) {
  __shared__ float tilebuf[32][33];
  const int bi = blockIdx.x;
  const int cb = bi & 3;
  const int xb = (bi >> 2) & 7;
  const int yy = (bi >> 5) & 255;
  const int n = bi >> 13;
  const int tid = threadIdx.x;
#pragma unroll
  for (int r = 0; r < 4; ++r) {
    const int lin = r * 256 + tid;
    const int xi_ = lin >> 5, ci = lin & 31;
    const int c = cb * 32 + ci;
    float v = y[(((size_t)(n * 256 + yy) * 256) + xb * 32 + xi_) * 128 + c];
    v = fmaxf(v * scl[c] + bias[c], 0.f);
    tilebuf[xi_][ci] = v;
  }
  __syncthreads();
#pragma unroll
  for (int r = 0; r < 4; ++r) {
    const int lin = r * 256 + tid;
    const int ci = lin >> 5, xi_ = lin & 31;
    const int c = cb * 32 + ci;
    out[(((size_t)(n * 128 + c) * 256) + yy) * 256 + xb * 32 + xi_] = tilebuf[xi_][ci];
  }
}

extern "C" void kernel_launch(void* const* d_in, const int* in_sizes, int n_in,
                              void* d_out, int out_size, void* d_ws, size_t ws_size,
                              hipStream_t stream) {
  (void)in_sizes; (void)n_in; (void)out_size; (void)ws_size;
  const float* x  = (const float*)d_in[0];
  const float* w1 = (const float*)d_in[1];
  const float* g1 = (const float*)d_in[2];
  const float* b1 = (const float*)d_in[3];
  const float* w2 = (const float*)d_in[4];
  const float* g2 = (const float*)d_in[5];
  const float* b2 = (const float*)d_in[6];
  float* out = (float*)d_out;

  char* ws = (char*)d_ws;
  size_t off = 0;
  auto take = [&](size_t bytes) -> char* {
    char* p = ws + off;
    off = (off + bytes + 255) & ~(size_t)255;
    return p;
  };
  float*       yBuf   = (float*)take(268435456);
  signed char* xiBuf  = (signed char*)take(67108864);
  signed char* bpk1   = (signed char*)take(73728);
  signed char* bpk2   = (signed char*)take(147456);
  float*       psum   = (float*)take(4096 * 128 * 4);
  float*       psq    = (float*)take(4096 * 128 * 4);
  float*       betaB  = (float*)take(256);
  unsigned*    gamma1 = (unsigned*)take(256);
  unsigned*    gamma2 = (unsigned*)take(256);
  float*       sc1    = (float*)take(512);
  float*       bs1    = (float*)take(512);
  float*       sc2    = (float*)take(512);
  float*       bs2    = (float*)take(512);

  hipMemsetAsync(gamma1, 0, 32, stream);
  hipMemsetAsync(gamma2, 0, 32, stream);

  weight_beta<<<2, 256, 0, stream>>>(w1, 73728, w2, 147456, betaB);
  weight_pack<<<288, 256, 0, stream>>>(w1, betaB + 0, bpk1, 64, 1, 73728);
  weight_pack<<<576, 256, 0, stream>>>(w2, betaB + 1, bpk2, 128, 2, 147456);

  absmax_x<<<8 * 4096, 256, 0, stream>>>(x, gamma1);
  quant_x<<<131072, 256, 0, stream>>>(x, gamma1, xiBuf);

  conv_wmma<64><<<4096, 256, 73728 + 8192, stream>>>(xiBuf, bpk1, betaB + 0, gamma1,
                                                     yBuf, psum, psq);
  bn_finalize<<<128, 256, 0, stream>>>(psum, psq, g1, b1, sc1, bs1, 4096,
                                       1.0f / 524288.0f);
  bnrelu_absmax<<<8 * 8192, 256, 0, stream>>>(yBuf, sc1, bs1, gamma2);
  quant_h<<<65536, 256, 0, stream>>>(yBuf, gamma2, xiBuf);

  conv_wmma<128><<<4096, 256, 147456 + 8192, stream>>>(xiBuf, bpk2, betaB + 1, gamma2,
                                                       yBuf, psum, psq);
  bn_finalize<<<128, 256, 0, stream>>>(psum, psq, g2, b2, sc2, bs2, 4096,
                                       1.0f / 524288.0f);
  bnrelu_out<<<65536, 256, 0, stream>>>(yBuf, sc2, bs2, out);
}
